// QuantLinearSim_80324478370236
// MI455X (gfx1250) — matrix operand order repaired and checked by reading, not compiled, not run on hardware
//
#include <hip/hip_runtime.h>
#include <stdint.h>

typedef __attribute__((ext_vector_type(16))) __bf16 v16bf;
typedef __attribute__((ext_vector_type(8)))  float  v8f;

#define T_DIM   4096   // rows of x2 (2*2048)
#define IN_DIM  4096
#define OUT_DIM 4096
#define NELEM   ((size_t)T_DIM * IN_DIM)
#define LDSP    40     // LDS row pitch in bf16 (80B -> conflict-free b128 frags)

static __device__ __forceinline__ unsigned short bfbits(__bf16 h) {
  return __builtin_bit_cast(unsigned short, h);
}

// ------------------------------------------------------------------
// Kernel 1: per-input-channel order statistics via LDS bitonic sort.
// ------------------------------------------------------------------
__global__ __launch_bounds__(256)
void stats_kernel(const float* __restrict__ x,
                  float* __restrict__ loA, float* __restrict__ upA,
                  float* __restrict__ offA, float* __restrict__ rngA) {
  __shared__ float s[T_DIM];
  __shared__ float rmin[256];
  __shared__ float rmax[256];
  const int c   = blockIdx.x;
  const int tid = threadIdx.x;

  for (int i = tid; i < T_DIM; i += 256)
    s[i] = x[(size_t)i * IN_DIM + c];

  for (int k = 2; k <= T_DIM; k <<= 1) {
    for (int j = k >> 1; j > 0; j >>= 1) {
      __syncthreads();
      for (int t = tid; t < T_DIM / 2; t += 256) {
        int i = ((t & ~(j - 1)) << 1) | (t & (j - 1));
        int p = i | j;
        float a = s[i], b = s[p];
        bool up = ((i & k) == 0);
        if ((a > b) == up) { s[i] = b; s[p] = a; }
      }
    }
  }
  __syncthreads();

  // quantile positions: 4095*0.0005 = 2.0475 ; 4095*0.9995 = 4092.9525
  const float lower = s[2]    + 0.0475f * (s[3]    - s[2]);
  const float upper = s[4092] + 0.9525f * (s[4093] - s[4092]);
  const float med   = s[(T_DIM - 1) / 2];

  float mn = __builtin_inff(), mx = -__builtin_inff();
  for (int i = tid; i < T_DIM; i += 256) {
    float v  = s[i];
    float tv = (v <= lower || v >= upper) ? med : v;
    mn = fminf(mn, tv);
    mx = fmaxf(mx, tv);
  }
  rmin[tid] = mn; rmax[tid] = mx;
  __syncthreads();
  for (int st = 128; st > 0; st >>= 1) {
    if (tid < st) {
      rmin[tid] = fminf(rmin[tid], rmin[tid + st]);
      rmax[tid] = fmaxf(rmax[tid], rmax[tid + st]);
    }
    __syncthreads();
  }
  if (tid == 0) {
    float minv = rmin[0], maxv = rmax[0];
    loA[c]  = lower;
    upA[c]  = upper;
    offA[c] = 0.5f * (maxv + minv);
    rngA[c] = 0.5f * (maxv - minv);
  }
}

// ------------------------------------------------------------------
// Kernel 2: NF4 fake-quantization -> packed bf16 hi/lo split arrays.
// ------------------------------------------------------------------
__device__ __forceinline__
float quant_one(float xv, float lo, float up, float of, float rg,
                const float* __restrict__ lv) {
  const bool masked = (xv <= lo) || (xv >= up);
  float qv;
  if (masked) {
    qv = (xv - of) + of;   // same rounding order as reference
  } else {
    float xsc = (xv - of) / rg;
    float bd = __builtin_inff(), bv = 0.0f;
#pragma unroll
    for (int k = 0; k < 16; ++k) {
      float d = fabsf(xsc - lv[k]);
      if (d < bd) { bd = d; bv = lv[k]; }
    }
    qv = bv * rg + of;
  }
  if (!__builtin_isfinite(qv)) qv = 0.0f;   // nan_to_num
  return qv;
}

__global__ __launch_bounds__(256)
void quant_kernel(const float* __restrict__ x, const float* __restrict__ lut,
                  const float* __restrict__ loA, const float* __restrict__ upA,
                  const float* __restrict__ offA, const float* __restrict__ rngA,
                  __bf16* __restrict__ qh, __bf16* __restrict__ ql) {
  const int i4 = blockIdx.x * 256 + threadIdx.x;       // float4 index
  const int c4 = i4 & (IN_DIM / 4 - 1);

  float lv[16];
#pragma unroll
  for (int k = 0; k < 16; ++k) lv[k] = lut[k];

  float4 xv = ((const float4*)x)[i4];
  float4 lo = ((const float4*)loA)[c4];
  float4 up = ((const float4*)upA)[c4];
  float4 of = ((const float4*)offA)[c4];
  float4 rg = ((const float4*)rngA)[c4];

  float o[4];
  o[0] = quant_one(xv.x, lo.x, up.x, of.x, rg.x, lv);
  o[1] = quant_one(xv.y, lo.y, up.y, of.y, rg.y, lv);
  o[2] = quant_one(xv.z, lo.z, up.z, of.z, rg.z, lv);
  o[3] = quant_one(xv.w, lo.w, up.w, of.w, rg.w, lv);

  unsigned hb[4], lb[4];
#pragma unroll
  for (int i = 0; i < 4; ++i) {
    __bf16 h = (__bf16)o[i];
    __bf16 l = (__bf16)(o[i] - (float)h);
    hb[i] = bfbits(h); lb[i] = bfbits(l);
  }
  uint2 ph = make_uint2(hb[0] | (hb[1] << 16), hb[2] | (hb[3] << 16));
  uint2 pl = make_uint2(lb[0] | (lb[1] << 16), lb[2] | (lb[3] << 16));
  *(uint2*)(qh + (size_t)i4 * 4) = ph;
  *(uint2*)(ql + (size_t)i4 * 4) = pl;
}

// ------------------------------------------------------------------
// Kernel 3: transpose + hi/lo split of W (K x N) -> Wt (N x K) bf16.
// 64x64 tiles through LDS for coalescing.
// ------------------------------------------------------------------
__global__ __launch_bounds__(256)
void wsplit_kernel(const float* __restrict__ W,
                   __bf16* __restrict__ wth, __bf16* __restrict__ wtl) {
  __shared__ float tile[64][65];
  const int tid = threadIdx.x;
  const int kb = blockIdx.y * 64;
  const int nb = blockIdx.x * 64;
#pragma unroll
  for (int it = 0; it < 4; ++it) {
    int t = tid + it * 256;
    int r = t >> 4;           // 0..63 (K within tile)
    int c = (t & 15) * 4;     // 0..60 (N within tile)
    float4 v = *(const float4*)(W + (size_t)(kb + r) * OUT_DIM + nb + c);
    tile[r][c] = v.x; tile[r][c + 1] = v.y; tile[r][c + 2] = v.z; tile[r][c + 3] = v.w;
  }
  __syncthreads();
#pragma unroll
  for (int it = 0; it < 4; ++it) {
    int t = tid + it * 256;
    int n = t >> 4;           // 0..63 (N within tile)
    int k = (t & 15) * 4;     // 0..60 (K within tile)
    unsigned hb[4], lb[4];
#pragma unroll
    for (int i = 0; i < 4; ++i) {
      float f = tile[k + i][n];
      __bf16 h = (__bf16)f;
      __bf16 l = (__bf16)(f - (float)h);
      hb[i] = bfbits(h); lb[i] = bfbits(l);
    }
    uint2 ph = make_uint2(hb[0] | (hb[1] << 16), hb[2] | (hb[3] << 16));
    uint2 pl = make_uint2(lb[0] | (lb[1] << 16), lb[2] | (lb[3] << 16));
    size_t o = (size_t)(nb + n) * IN_DIM + kb + k;
    *(uint2*)(wth + o) = ph;
    *(uint2*)(wtl + o) = pl;
  }
}

// ------------------------------------------------------------------
// Kernel 4: bf16x3 split-precision GEMM with async-to-LDS double
// buffering. y = q @ W via v_wmma_f32_16x16x32_bf16.
// Per chunk: 8 global_load_async_to_lds_b128 issues/wave, 24 WMMAs.
// ------------------------------------------------------------------
union FragBf { v16bf v; uint4 u[2]; };

__global__ __launch_bounds__(256)
void gemm_kernel(const __bf16* __restrict__ Qh, const __bf16* __restrict__ Ql,
                 const __bf16* __restrict__ Wh, const __bf16* __restrict__ Wl,
                 float* __restrict__ Y) {
  // smem[buf][arr][row][k] : arr 0=Ah 1=Al 2=Bh 3=Bl
  __shared__ __bf16 smem[2][4][128][LDSP];

  const int tid  = threadIdx.x;
  const int lane = tid & 31;
  const int wave = tid >> 5;
  const int m0 = blockIdx.y * 128;
  const int n0 = blockIdx.x * 128;
  const int wm = (wave & 3) * 32;
  const int wn = (wave >> 2) * 64;
  const int lr = lane & 15;
  const int lh = lane >> 4;

  v8f acc[2][4] = {};

  // issue one K-chunk's copies (4 arrays * 128 rows * 4 x 16B segments
  // = 2048 B128 ops; 8 per thread)
  auto issue = [&](int buf, int k0) {
#pragma unroll
    for (int it = 0; it < 8; ++it) {
      const int arr = it >> 1;
      const int sub = ((it & 1) << 8) + tid;   // 0..511
      const int r   = sub >> 2;                // 0..127
      const int seg = sub & 3;                 // 16B segment
      const __bf16* gb;
      int rowbase;
      if (arr == 0)      { gb = Qh; rowbase = m0; }
      else if (arr == 1) { gb = Ql; rowbase = m0; }
      else if (arr == 2) { gb = Wh; rowbase = n0; }
      else               { gb = Wl; rowbase = n0; }
      uint64_t ga = (uint64_t)(uintptr_t)(gb + (size_t)(rowbase + r) * IN_DIM + k0 + seg * 8);
      unsigned  la = (unsigned)(uintptr_t)&smem[buf][arr][r][seg * 8];
      asm volatile("global_load_async_to_lds_b128 %0, %1, off"
                   :: "v"(la), "v"(ga) : "memory");
    }
  };

  issue(0, 0);

  const int NCHUNK = IN_DIM / 32;
  for (int kc = 0; kc < NCHUNK; ++kc) {
    const int buf = kc & 1;
    if (kc + 1 < NCHUNK) {
      issue(buf ^ 1, (kc + 1) * 32);
      asm volatile("s_wait_asynccnt 0x8" ::: "memory");   // current buffer done
    } else {
      asm volatile("s_wait_asynccnt 0x0" ::: "memory");
    }
    __syncthreads();

    // fragments (documented 16-bit A 16x32 / B 32x16 VGPR layouts)
    FragBf fAh[2], fAl[2], fBh[4], fBl[4];
#pragma unroll
    for (int mt = 0; mt < 2; ++mt) {
      int r  = wm + mt * 16 + lr;
      int e0 = lh * 8;
      fAh[mt].u[0] = *(const uint4*)&smem[buf][0][r][e0];
      fAh[mt].u[1] = *(const uint4*)&smem[buf][0][r][e0 + 16];
      fAl[mt].u[0] = *(const uint4*)&smem[buf][1][r][e0];
      fAl[mt].u[1] = *(const uint4*)&smem[buf][1][r][e0 + 16];
    }
#pragma unroll
    for (int nt = 0; nt < 4; ++nt) {
      int r  = wn + nt * 16 + lr;
      int e0 = lh * 8;
      fBh[nt].u[0] = *(const uint4*)&smem[buf][2][r][e0];
      fBh[nt].u[1] = *(const uint4*)&smem[buf][2][r][e0 + 16];
      fBl[nt].u[0] = *(const uint4*)&smem[buf][3][r][e0];
      fBl[nt].u[1] = *(const uint4*)&smem[buf][3][r][e0 + 16];
    }

#pragma unroll
    for (int mt = 0; mt < 2; ++mt)
#pragma unroll
      for (int nt = 0; nt < 4; ++nt) {
        acc[mt][nt] = __builtin_amdgcn_wmma_f32_16x16x32_bf16(
            false, fAh[mt].v, false, fBh[nt].v, (short)0, acc[mt][nt], false, false);
        acc[mt][nt] = __builtin_amdgcn_wmma_f32_16x16x32_bf16(
            false, fAl[mt].v, false, fBh[nt].v, (short)0, acc[mt][nt], false, false);
        acc[mt][nt] = __builtin_amdgcn_wmma_f32_16x16x32_bf16(
            false, fAh[mt].v, false, fBl[nt].v, (short)0, acc[mt][nt], false, false);
      }
    __syncthreads();
  }

  // epilogue: C/D layout VGPR v, lane l -> M = v + 8*(l>>4), N = l&15
#pragma unroll
  for (int mt = 0; mt < 2; ++mt) {
#pragma unroll
    for (int nt = 0; nt < 4; ++nt) {
      int mbase = m0 + wm + mt * 16 + lh * 8;
      int ncol  = n0 + wn + nt * 16 + lr;
#pragma unroll
      for (int vv = 0; vv < 8; ++vv)
        Y[(size_t)(mbase + vv) * OUT_DIM + ncol] = acc[mt][nt][vv];
    }
  }
}

// ------------------------------------------------------------------
extern "C" void kernel_launch(void* const* d_in, const int* in_sizes, int n_in,
                              void* d_out, int out_size, void* d_ws, size_t ws_size,
                              hipStream_t stream) {
  (void)in_sizes; (void)n_in; (void)out_size; (void)ws_size;
  const float* x   = (const float*)d_in[0];   // (2,2048,4096) f32
  const float* w   = (const float*)d_in[1];   // (4096,4096)   f32 (in,out)
  const float* lut = (const float*)d_in[2];   // (16,)         f32
  float* out = (float*)d_out;

  // workspace layout: 4 bf16 planes (32MB each) + stats
  __bf16* qh  = (__bf16*)d_ws;
  __bf16* ql  = qh + NELEM;
  __bf16* wth = ql + NELEM;
  __bf16* wtl = wth + NELEM;
  float* loA  = (float*)(wtl + NELEM);
  float* upA  = loA + IN_DIM;
  float* offA = upA + IN_DIM;
  float* rngA = offA + IN_DIM;

  stats_kernel<<<IN_DIM, 256, 0, stream>>>(x, loA, upA, offA, rngA);

  const int total4 = (T_DIM * IN_DIM) / 4;
  quant_kernel<<<total4 / 256, 256, 0, stream>>>(x, lut, loA, upA, offA, rngA, qh, ql);

  wsplit_kernel<<<dim3(OUT_DIM / 64, IN_DIM / 64), 256, 0, stream>>>(w, wth, wtl);

  gemm_kernel<<<dim3(OUT_DIM / 128, T_DIM / 128), 256, 0, stream>>>(qh, ql, wth, wtl, out);
}
